// DynamicFieldPropagator_23089744183586
// MI455X (gfx1250) — matrix operand order repaired and checked by reading, not compiled, not run on hardware
//
#include <hip/hip_runtime.h>

#define B_ 4
#define N_ 2048
#define D_ 256
#define H_ 8
#define DH_ 32
#define NSTEPS 4

typedef __attribute__((ext_vector_type(16))) __bf16 v16bf;
typedef __attribute__((ext_vector_type(8)))  float  v8f;

union BF16Tile { v16bf v; uint4 q[2]; unsigned int u[8]; };

#define SCALE_ 0.17677669529663687f   /* 1/sqrt(Dh) */

__device__ __forceinline__ unsigned short f2bf(float x) {
    __bf16 h = (__bf16)x;                       // native cvt (v_cvt_pk_bf16_f32 when paired)
    return __builtin_bit_cast(unsigned short, h);
}
__device__ __forceinline__ unsigned int pack2(float lo, float hi) {
    return (unsigned int)f2bf(lo) | ((unsigned int)f2bf(hi) << 16);
}

// ---------------------------------------------------------------------------
// K0: fp32 -> bf16 pack of Wi and Wo (row-major [d][k], k contiguous)
// ---------------------------------------------------------------------------
__global__ void pack_weights(const float* __restrict__ Wi, const float* __restrict__ Wo,
                             unsigned short* __restrict__ Wib, unsigned short* __restrict__ Wob) {
    int i = blockIdx.x * blockDim.x + threadIdx.x;
    if (i < D_ * D_) {
        Wib[i] = f2bf(Wi[i]);
        Wob[i] = f2bf(Wo[i]);
    }
}

// ---------------------------------------------------------------------------
// K1: pack f fp32 [B,N,D] -> fb bf16 [B,N,D], fQb = bf16(f*scale) [B,N,D],
//     fTb bf16 [B,D,N]
// ---------------------------------------------------------------------------
__global__ void pack_fields(const float* __restrict__ f,
                            unsigned short* __restrict__ fb,
                            unsigned short* __restrict__ fQb,
                            unsigned short* __restrict__ fTb) {
    int i = blockIdx.x * blockDim.x + threadIdx.x;       // 0 .. B*N*D-1 (2^21)
    int d = i & (D_ - 1);
    int t = (i >> 8) & (N_ - 1);
    int b = i >> 19;
    float v = f[i];
    fb[i]  = f2bf(v);
    fQb[i] = f2bf(v * SCALE_);
    fTb[((long)(b * D_ + d) << 11) + t] = f2bf(v);
}

// ---------------------------------------------------------------------------
// K2: flash-style interference, LDS-free.
//   simT tile = K(16x32) x Q^T(32x16) via v_wmma_f32_16x16x32_bf16
//   (scale folded into pre-scaled Q copy). Nonlinearity:
//   I = a*s + b*|s| + w2*cos(s)  with a=(w0+w1)/2, b=(w0-w1)/2.
//   simT D-layout == I^T B-operand layout up to a half-wave swap, done with
//   __shfl_xor(.,16) + cndmask in registers (no LDS round trip).
//   O^T accumulated as V^T(16x32) x I^T(32x16) WMMAs. sim never hits memory.
// ---------------------------------------------------------------------------
__global__ __launch_bounds__(128) void interference_kernel(
        const unsigned short* __restrict__ fb,   // [B,N,D] bf16
        const unsigned short* __restrict__ fQb,  // [B,N,D] bf16, pre-scaled
        const unsigned short* __restrict__ fTb,  // [B,D,N] bf16
        const float* __restrict__ wts,           // [3]
        unsigned short* __restrict__ Ob)         // [B,N,D] bf16 out
{
    const int lane = threadIdx.x & 31;
    const int w    = threadIdx.x >> 5;
    const int id   = blockIdx.x * 4 + w;          // 0..4095
    const int qt   = id & 127;
    const int h    = (id >> 7) & 7;
    const int b    = id >> 10;
    const int q0   = qt << 4;

    const float w0 = wts[0], w1 = wts[1], w2 = wts[2];
    const float ca = 0.5f * (w0 + w1);            // coeff of s
    const float cb = 0.5f * (w0 - w1);            // coeff of |s|

    const int hi  = lane >> 4;                    // 0/1 lane half
    const int l15 = lane & 15;

    // Q^T B-operand (pre-scaled): lane = q column, dh pairs contiguous.
    BF16Tile Qs;
    {
        long e = (long)(b * N_ + q0 + l15) * D_ + h * DH_;
        const uint4* p = (const uint4*)(fQb + e);
        Qs.q[0] = p[hi];
        Qs.q[1] = p[hi + 2];
    }

    v8f o0 = {};   // O^T tile, dh rows 0..15
    v8f o1 = {};   // O^T tile, dh rows 16..31

    for (int m0 = 0; m0 < N_; m0 += 32) {
        // --- A operands: K rows m0..m0+15 (tile0) and m0+16..m0+31 (tile1) ---
        BF16Tile K0, K1;
        {
            long e0 = (long)(b * N_ + m0 + l15) * D_ + h * DH_;
            const uint4* p0 = (const uint4*)(fb + e0);
            K0.q[0] = p0[hi]; K0.q[1] = p0[hi + 2];
            const uint4* p1 = (const uint4*)(fb + e0 + 16l * D_);
            K1.q[0] = p1[hi]; K1.q[1] = p1[hi + 2];
        }
        v8f z = {};
        v8f c0 = __builtin_amdgcn_wmma_f32_16x16x32_bf16(false, K0.v, false, Qs.v, (short)0, z, false, false);
        v8f c1 = __builtin_amdgcn_wmma_f32_16x16x32_bf16(false, K1.v, false, Qs.v, (short)0, z, false, false);

        // --- nonlinearity (already scaled) + pack k-pairs along m (in-lane) ---
        unsigned int t0u[4], t1u[4];
        #pragma unroll
        for (int i = 0; i < 4; ++i) {
            float s0 = c0[2 * i],     s1 = c0[2 * i + 1];
            float i0 = fmaf(ca, s0, fmaf(cb, fabsf(s0), w2 * __cosf(s0)));
            float i1 = fmaf(ca, s1, fmaf(cb, fabsf(s1), w2 * __cosf(s1)));
            t0u[i] = pack2(i0, i1);
            s0 = c1[2 * i];  s1 = c1[2 * i + 1];
            i0 = fmaf(ca, s0, fmaf(cb, fabsf(s0), w2 * __cosf(s0)));
            i1 = fmaf(ca, s1, fmaf(cb, fabsf(s1), w2 * __cosf(s1)));
            t1u[i] = pack2(i0, i1);
        }
        // --- half-wave swap builds the I^T B-operand (k = m0..m0+31) ---
        BF16Tile Bi;
        #pragma unroll
        for (int i = 0; i < 4; ++i) {
            unsigned int sw0 = (unsigned int)__shfl_xor((int)t0u[i], 16, 32);
            unsigned int sw1 = (unsigned int)__shfl_xor((int)t1u[i], 16, 32);
            Bi.u[i]     = hi ? sw1   : t0u[i];
            Bi.u[4 + i] = hi ? t1u[i] : sw0;
        }
        // --- A' = V^T chunks from fTb: row = dh, K = m contiguous ---
        BF16Tile V0, V1;
        {
            long e0 = ((long)(b * D_ + h * DH_ + l15) << 11) + m0;
            const uint4* p0 = (const uint4*)(fTb + e0);
            V0.q[0] = p0[hi * 2]; V0.q[1] = p0[hi * 2 + 1];
            const uint4* p1 = (const uint4*)(fTb + e0 + (16l << 11));
            V1.q[0] = p1[hi * 2]; V1.q[1] = p1[hi * 2 + 1];
        }
        o0 = __builtin_amdgcn_wmma_f32_16x16x32_bf16(false, V0.v, false, Bi.v, (short)0, o0, false, false);
        o1 = __builtin_amdgcn_wmma_f32_16x16x32_bf16(false, V1.v, false, Bi.v, (short)0, o1, false, false);
    }

    // --- store O (lane n holds O[token q0+n][dh..dh+7]) as bf16, one b128 per tile ---
    {
        long t = (long)(b * N_ + q0 + l15);
        int dbase = h * DH_ + hi * 8;
        uint4 st;
        st.x = pack2(o0[0], o0[1]); st.y = pack2(o0[2], o0[3]);
        st.z = pack2(o0[4], o0[5]); st.w = pack2(o0[6], o0[7]);
        *(uint4*)(Ob + t * D_ + dbase) = st;
        st.x = pack2(o1[0], o1[1]); st.y = pack2(o1[2], o1[3]);
        st.z = pack2(o1[4], o1[5]); st.w = pack2(o1[6], o1[7]);
        *(uint4*)(Ob + t * D_ + dbase + 16) = st;
    }
}

// ---------------------------------------------------------------------------
// K3: fused (O @ Wi^T + bi)*coupling + alpha*laplacian + Euler update.
// One wave = one 16-token x 16-d output tile, 8 WMMAs over K=256.
// Writes f_next fp32 plus bf16 packs (fb, fQb, fTb) for the next step.
// ---------------------------------------------------------------------------
__global__ __launch_bounds__(128) void project_step_kernel(
        const unsigned short* __restrict__ Ob,
        const unsigned short* __restrict__ Wib,
        const float* __restrict__ bi,
        const float* __restrict__ coupling,
        const float* __restrict__ dcoeff,
        const float* __restrict__ fcur,
        float* __restrict__ fnext,
        unsigned short* __restrict__ fb,
        unsigned short* __restrict__ fQb,
        unsigned short* __restrict__ fTb)
{
    const int lane = threadIdx.x & 31;
    const int w    = threadIdx.x >> 5;
    const int id   = blockIdx.x * 4 + w;          // 0..8191
    const int d0   = (id & 15) << 4;
    const int rt   = id >> 4;                     // row tile across B*N
    const int b    = rt >> 7;
    const int t0   = (rt & 127) << 4;
    const int hi   = lane >> 4, l15 = lane & 15;

    v8f acc = {};
    const uint4* pa = (const uint4*)(Ob  + (long)(b * N_ + t0 + l15) * D_);
    const uint4* pb = (const uint4*)(Wib + (long)(d0 + l15) * D_);
    #pragma unroll
    for (int k0 = 0; k0 < D_; k0 += 32) {
        BF16Tile Aa, Bb;
        int qa = (k0 >> 3) + hi;
        Aa.q[0] = pa[qa]; Aa.q[1] = pa[qa + 2];
        int qb = (k0 >> 3) + hi;
        Bb.q[0] = pb[qb]; Bb.q[1] = pb[qb + 2];
        acc = __builtin_amdgcn_wmma_f32_16x16x32_bf16(false, Aa.v, false, Bb.v, (short)0, acc, false, false);
    }

    const float alpha = fminf(fmaxf(dcoeff[0], 0.01f), 1.0f);
    const int d = d0 + l15;
    const float cpl = fminf(fmaxf(coupling[d], 0.01f), 10.0f);
    const float bid = bi[d];
    const float dt = 0.01f;

    #pragma unroll
    for (int j = 0; j < 8; ++j) {
        int t = t0 + hi * 8 + j;
        long idx = (long)(b * N_ + t) * D_ + d;
        float fc = fcur[idx];
        float lap;
        if (t == 0)            lap = fcur[idx + D_] - fc;
        else if (t == N_ - 1)  lap = fcur[idx - D_] - fc;
        else                   lap = fcur[idx + D_] - 2.f * fc + fcur[idx - D_];
        float interf = cpl * (acc[j] + bid);
        float fn = fc + dt * (alpha * lap + 0.5f * interf);
        fnext[idx] = fn;
        fb[idx]  = f2bf(fn);
        fQb[idx] = f2bf(fn * SCALE_);
        fTb[((long)(b * D_ + d) << 11) + t] = f2bf(fn);
    }
}

// ---------------------------------------------------------------------------
// K4: final f @ Wo^T + bo -> d_out (fp32)
// ---------------------------------------------------------------------------
__global__ __launch_bounds__(128) void final_proj_kernel(
        const unsigned short* __restrict__ fb,
        const unsigned short* __restrict__ Wob,
        const float* __restrict__ bo,
        float* __restrict__ out)
{
    const int lane = threadIdx.x & 31;
    const int w    = threadIdx.x >> 5;
    const int id   = blockIdx.x * 4 + w;
    const int d0   = (id & 15) << 4;
    const int rt   = id >> 4;
    const int b    = rt >> 7;
    const int t0   = (rt & 127) << 4;
    const int hi   = lane >> 4, l15 = lane & 15;

    v8f acc = {};
    const uint4* pa = (const uint4*)(fb  + (long)(b * N_ + t0 + l15) * D_);
    const uint4* pb = (const uint4*)(Wob + (long)(d0 + l15) * D_);
    #pragma unroll
    for (int k0 = 0; k0 < D_; k0 += 32) {
        BF16Tile Aa, Bb;
        int qa = (k0 >> 3) + hi;
        Aa.q[0] = pa[qa]; Aa.q[1] = pa[qa + 2];
        int qb = (k0 >> 3) + hi;
        Bb.q[0] = pb[qb]; Bb.q[1] = pb[qb + 2];
        acc = __builtin_amdgcn_wmma_f32_16x16x32_bf16(false, Aa.v, false, Bb.v, (short)0, acc, false, false);
    }
    const int d = d0 + l15;
    const float bod = bo[d];
    #pragma unroll
    for (int j = 0; j < 8; ++j) {
        int t = t0 + hi * 8 + j;
        out[(long)(b * N_ + t) * D_ + d] = acc[j] + bod;
    }
}

// ---------------------------------------------------------------------------
extern "C" void kernel_launch(void* const* d_in, const int* in_sizes, int n_in,
                              void* d_out, int out_size, void* d_ws, size_t ws_size,
                              hipStream_t stream) {
    (void)in_sizes; (void)n_in; (void)out_size; (void)ws_size;
    const float* f_in     = (const float*)d_in[0];
    const float* dcoeff   = (const float*)d_in[1];
    const float* coupling = (const float*)d_in[2];
    const float* wts      = (const float*)d_in[3];
    const float* Wi       = (const float*)d_in[4];
    const float* bi       = (const float*)d_in[5];
    const float* Wo       = (const float*)d_in[6];
    const float* bo       = (const float*)d_in[7];

    char* ws = (char*)d_ws;
    size_t off = 0;
    float*          f0  = (float*)(ws + off);          off += (size_t)B_ * N_ * D_ * 4;
    float*          f1  = (float*)(ws + off);          off += (size_t)B_ * N_ * D_ * 4;
    unsigned short* fb  = (unsigned short*)(ws + off); off += (size_t)B_ * N_ * D_ * 2;
    unsigned short* fQb = (unsigned short*)(ws + off); off += (size_t)B_ * N_ * D_ * 2;
    unsigned short* fTb = (unsigned short*)(ws + off); off += (size_t)B_ * N_ * D_ * 2;
    unsigned short* Ob  = (unsigned short*)(ws + off); off += (size_t)B_ * N_ * D_ * 2;
    unsigned short* Wib = (unsigned short*)(ws + off); off += (size_t)D_ * D_ * 2;
    unsigned short* Wob = (unsigned short*)(ws + off); off += (size_t)D_ * D_ * 2;

    pack_weights<<<(D_ * D_ + 255) / 256, 256, 0, stream>>>(Wi, Wo, Wib, Wob);
    pack_fields<<<(B_ * N_ * D_) / 256, 256, 0, stream>>>(f_in, fb, fQb, fTb);

    const float* fcur = f_in;
    float* fnext = f0;
    for (int s = 0; s < NSTEPS; ++s) {
        interference_kernel<<<1024, 128, 0, stream>>>(fb, fQb, fTb, wts, Ob);
        project_step_kernel<<<2048, 128, 0, stream>>>(Ob, Wib, bi, coupling, dcoeff,
                                                      fcur, fnext, fb, fQb, fTb);
        fcur  = fnext;
        fnext = (fnext == f0) ? f1 : f0;
    }
    final_proj_kernel<<<2048, 128, 0, stream>>>(fb, Wob, bo, (float*)d_out);
}
